// compute_G_2_11768210391386
// MI455X (gfx1250) — matrix operand order repaired and checked by reading, not compiled, not run on hardware
//
#include <hip/hip_runtime.h>

#define NN 8192
#define EE 4096
#define BM 128
#define BN 128
#define BK 64
#define LDA 72    // padded LDS stride (elements) for A tile rows (128 x 64)
#define LDB 136   // padded LDS stride (elements) for B tile rows (64 x 128)
#define BUFBIT 32768u   // byte distance between double buffers (power of two)

typedef __attribute__((ext_vector_type(16))) __bf16 v16bf;
typedef __attribute__((ext_vector_type(8)))  float  v8f;

union Frag {
    uint4 u[2];
    v16bf v;
};

__device__ __forceinline__ unsigned short f2bf(float f) {
    unsigned int u = __float_as_uint(f);
    u += 0x7FFFu + ((u >> 16) & 1u);       // round-to-nearest-even
    return (unsigned short)(u >> 16);
}

// Async copy: 16 bytes/lane global -> LDS, no VGPR staging (ASYNCcnt path).
// GVS addressing: mem = SADDR(64b, SGPR) + VADDR(32b, VGPR) ; VDST = LDS byte addr.
__device__ __forceinline__ void async_load_b128(unsigned lds_addr, unsigned voff,
                                                const void* sbase) {
    asm volatile("global_load_async_to_lds_b128 %0, %1, %2 offset:0"
                 :
                 : "v"(lds_addr), "v"(voff), "s"(sbase)
                 : "memory");
}

__device__ __forceinline__ void wait_async0() {
    asm volatile("s_wait_asynccnt 0x0" ::: "memory");
}

// ---------------------------------------------------------------------------
// Kernel 1: d[n] = rsqrt( sum_e H[n,e] * W[e] )     (one block per row)
// ---------------------------------------------------------------------------
__global__ void degree_rsqrt_kernel(const float* __restrict__ H,
                                    const float* __restrict__ W,
                                    float* __restrict__ d) {
    const int n = blockIdx.x;
    const int t = threadIdx.x;
    const float4* __restrict__ Hrow = (const float4*)(H + (size_t)n * EE);
    const float4* __restrict__ W4   = (const float4*)W;
    float s = 0.0f;
#pragma unroll
    for (int j = 0; j < 4; ++j) {          // EE/4 float4s = 1024, 256 threads
        int i4 = t + j * 256;
        float4 h = Hrow[i4];
        float4 w = W4[i4];
        s += h.x * w.x + h.y * w.y + h.z * w.z + h.w * w.w;
    }
    __shared__ float red[256];
    red[t] = s;
    __syncthreads();
    for (int off = 128; off > 0; off >>= 1) {
        if (t < off) red[t] += red[t + off];
        __syncthreads();
    }
    if (t == 0) d[n] = rsqrtf(red[0]);
}

// ---------------------------------------------------------------------------
// Kernel 2a: Abf[n,e] = bf16( d[n] * H[n,e] )        (row-major [NN x EE])
// ---------------------------------------------------------------------------
__global__ void scaleA_kernel(const float* __restrict__ H,
                              const float* __restrict__ d,
                              unsigned short* __restrict__ Abf) {
    const size_t total  = (size_t)NN * EE / 8;           // vec8 chunks
    const size_t stride = (size_t)gridDim.x * blockDim.x;
    for (size_t i = (size_t)blockIdx.x * blockDim.x + threadIdx.x;
         i < total; i += stride) {
        size_t e8 = i * 8;
        int n = (int)(e8 / EE);                          // EE % 8 == 0
        float dn = d[n];
        const float4* src = (const float4*)(H + e8);
        float4 a = src[0], b = src[1];
        uint4 o;
        o.x = (unsigned)f2bf(a.x * dn) | ((unsigned)f2bf(a.y * dn) << 16);
        o.y = (unsigned)f2bf(a.z * dn) | ((unsigned)f2bf(a.w * dn) << 16);
        o.z = (unsigned)f2bf(b.x * dn) | ((unsigned)f2bf(b.y * dn) << 16);
        o.w = (unsigned)f2bf(b.z * dn) | ((unsigned)f2bf(b.w * dn) << 16);
        ((uint4*)Abf)[i] = o;
    }
}

// ---------------------------------------------------------------------------
// Kernel 2b: Bbf[e,n] = bf16( W[e] * invDE_HT[e,n] * d[n] )   ([EE x NN])
// ---------------------------------------------------------------------------
__global__ void scaleB_kernel(const float* __restrict__ R,
                              const float* __restrict__ W,
                              const float* __restrict__ d,
                              unsigned short* __restrict__ Bbf) {
    const size_t total  = (size_t)EE * NN / 8;
    const size_t stride = (size_t)gridDim.x * blockDim.x;
    for (size_t i = (size_t)blockIdx.x * blockDim.x + threadIdx.x;
         i < total; i += stride) {
        size_t idx = i * 8;
        int e = (int)(idx / NN);                         // NN % 8 == 0
        int n = (int)(idx % NN);
        float we = W[e];
        const float4* src = (const float4*)(R + idx);
        const float4* dv  = (const float4*)(d + n);
        float4 a = src[0], b = src[1];
        float4 d0 = dv[0], d1 = dv[1];
        uint4 o;
        o.x = (unsigned)f2bf(a.x * we * d0.x) | ((unsigned)f2bf(a.y * we * d0.y) << 16);
        o.y = (unsigned)f2bf(a.z * we * d0.z) | ((unsigned)f2bf(a.w * we * d0.w) << 16);
        o.z = (unsigned)f2bf(b.x * we * d1.x) | ((unsigned)f2bf(b.y * we * d1.y) << 16);
        o.w = (unsigned)f2bf(b.z * we * d1.z) | ((unsigned)f2bf(b.w * we * d1.w) << 16);
        ((uint4*)Bbf)[i] = o;
    }
}

// ---------------------------------------------------------------------------
// Kernel 3: G = Abf @ Bbf   (bf16 WMMA, f32 accumulate)
// 256 threads (8 waves), tile 128x128, BK=64 (two 32-deep phases / barrier).
// Double-buffered LDS fed by global_load_async_to_lds_b128 (no staging VGPRs).
// Buffer flip is a register XOR (BUFBIT); K-loop has ONE body (unroll 1) so
// accumulators stay pinned. Waves 0-3 fetch the A tile (128x64), waves 4-7
// fetch the B tile (64x128): 8 async b128 per wave per K-step, uniform SGPR
// base pointer per role. Wave (wm,wn) owns a 32x64 subtile -> 16 WMMAs/step.
// LDS layout (bytes): A0 @0, A1 @32768 (18432 used each)
//                     B0 @65536, B1 @98304 (17408 used each)   total 128 KB
// launch_bounds(256,2): two blocks per WGP (256 KB LDS) so one block's
// compute covers the other's barrier/async-wait latency.
// ---------------------------------------------------------------------------
__global__ void __launch_bounds__(256, 2)
gemm_wmma_bf16_kernel(const unsigned short* __restrict__ A,   // [NN x EE] bf16
                      const unsigned short* __restrict__ B,   // [EE x NN] bf16
                      float* __restrict__ G) {                // [NN x NN] f32
    __shared__ unsigned char lds[4 * BUFBIT];

    const int t    = threadIdx.x;
    const int lane = t & 31;
    const int wave = t >> 5;          // 0..7
    const int wm   = wave >> 1;       // 0..3 : 32-row slab
    const int wn   = wave & 1;        // 0..1 : 64-col slab
    const int bm   = blockIdx.x * BM;
    const int bn   = blockIdx.y * BN;

    // ---- per-lane async-copy addresses (K-invariant; buffer bit XORed) ----
    unsigned ldso[8];                 // LDS byte address (raw, for asm VDST)
    unsigned voff[8];                 // global byte offset (GVS VADDR)
    const bool isA = (wave < 4);
#pragma unroll
    for (int i = 0; i < 8; ++i) {
        if (isA) {
            int c   = (wave * 8 + i) * 32 + lane;        // 0..1023 A chunks
            int row = c >> 3, col = (c & 7) << 3;        // 128 rows x 8 chunks
            voff[i] = (unsigned)(((unsigned)row * EE + col) * 2u);
            ldso[i] = (unsigned)(uintptr_t)(lds + (row * LDA + col) * 2);
        } else {
            int c   = ((wave - 4) * 8 + i) * 32 + lane;  // 0..1023 B chunks
            int row = c >> 4, col = (c & 15) << 3;       // 64 rows x 16 chunks
            voff[i] = (unsigned)(((unsigned)row * NN + col) * 2u);
            ldso[i] = (unsigned)(uintptr_t)(lds + 2 * BUFBIT + (row * LDB + col) * 2);
        }
    }
    // Uniform (SGPR) base pointers for the async SADDR; one per role.
    const unsigned short* baseA = A + (size_t)bm * EE;
    const unsigned short* baseB = B + bn;

    v8f acc[2][4];
#pragma unroll
    for (int mt = 0; mt < 2; ++mt)
#pragma unroll
        for (int nt = 0; nt < 4; ++nt)
            acc[mt][nt] = (v8f){0.f, 0.f, 0.f, 0.f, 0.f, 0.f, 0.f, 0.f};

    // One 32-deep compute phase.
    auto compute = [&](unsigned aoff, unsigned boff) {
        Frag a[2], b[4];
#pragma unroll
        for (int mt = 0; mt < 2; ++mt) {
            // A 16x32 bf16 frag: lane = M row; elems K[klo..+7], K[klo+16..+23]
            const unsigned char* p = lds + aoff + mt * (16 * LDA * 2);
            a[mt].u[0] = *(const uint4*)(p);
            a[mt].u[1] = *(const uint4*)(p + 32);
        }
#pragma unroll
        for (int nt = 0; nt < 4; ++nt) {
            // B 32x16 bf16 frag: lane = N col; elems K[klo..+7], K[klo+16..+23]
            const unsigned char* p = lds + boff + nt * 32;
#pragma unroll
            for (int s = 0; s < 16; ++s)
                b[nt].v[s] = *(const __bf16*)(p + ((s < 8) ? s : s + 8) * (LDB * 2));
        }
#pragma unroll
        for (int nt = 0; nt < 4; ++nt)
#pragma unroll
            for (int mt = 0; mt < 2; ++mt)
                acc[mt][nt] = __builtin_amdgcn_wmma_f32_16x16x32_bf16(
                    false, a[mt].v, false, b[nt].v,
                    (short)0, acc[mt][nt], false, false);
    };

    // prologue: fill buffer 0, then point async dests at buffer 1
    if (isA) {
#pragma unroll
        for (int i = 0; i < 8; ++i) async_load_b128(ldso[i], voff[i], baseA);
    } else {
#pragma unroll
        for (int i = 0; i < 8; ++i) async_load_b128(ldso[i], voff[i], baseB);
    }
    baseA += BK;
    baseB += (size_t)BK * NN;
#pragma unroll
    for (int i = 0; i < 8; ++i) ldso[i] ^= BUFBIT;

    const int KT   = EE / BK;                 // 64 K-steps
    const int klo  = (lane < 16) ? 0 : 8;     // A-frag K base per ISA layout
    const int mrow = lane & 15;
    unsigned offA = (unsigned)(((wm * 32 + mrow) * LDA + klo) * 2);
    unsigned offB = (unsigned)(2 * BUFBIT + (klo * LDB + wn * 64 + mrow) * 2);

#pragma unroll 1
    for (int kt = 0; kt < KT; ++kt) {
        wait_async0();                 // my tile-chunk loads for `cur` landed
        __syncthreads();               // everyone's landed; other buffer free
        if (kt + 1 < KT) {
            if (isA) {
#pragma unroll
                for (int i = 0; i < 8; ++i)
                    async_load_b128(ldso[i], voff[i], baseA);
            } else {
#pragma unroll
                for (int i = 0; i < 8; ++i)
                    async_load_b128(ldso[i], voff[i], baseB);
            }
            baseA += BK;
            baseB += (size_t)BK * NN;
        }

        compute(offA, offB);                                 // k-slice [0,32)
        compute(offA + 64, offB + 32 * (LDB * 2));           // k-slice [32,64)

        // flip double buffers (register XOR, keeps loop body unique)
#pragma unroll
        for (int i = 0; i < 8; ++i) ldso[i] ^= BUFBIT;
        offA ^= BUFBIT;
        offB ^= BUFBIT;
    }

    // Epilogue: C/D layout — VGPR r: lanes 0-15 => (M=r, N=lane),
    //                               lanes 16-31 => (M=r+8, N=lane-16).
    // One base address per lane; all 64 stores use constant element offsets.
    const int nc  = lane & 15;
    const int rhi = (lane >> 4) << 3;
    float* gp = G + (size_t)(bm + wm * 32 + rhi) * NN + (bn + wn * 64 + nc);
#pragma unroll
    for (int mt = 0; mt < 2; ++mt)
#pragma unroll
        for (int nt = 0; nt < 4; ++nt)
#pragma unroll
            for (int r = 0; r < 8; ++r)
                gp[(size_t)(mt * 16 + r) * NN + nt * 16] = acc[mt][nt][r];
}

// ---------------------------------------------------------------------------
// Host launcher
// Workspace layout: [ d : NN f32 (32 KB) | Abf : NN*EE bf16 | Bbf : EE*NN bf16 ]
// total ~134.3 MB.
// ---------------------------------------------------------------------------
extern "C" void kernel_launch(void* const* d_in, const int* in_sizes, int n_in,
                              void* d_out, int out_size, void* d_ws, size_t ws_size,
                              hipStream_t stream) {
    (void)in_sizes; (void)n_in; (void)out_size; (void)ws_size;
    const float* H        = (const float*)d_in[0];   // [NN x EE]
    const float* invDE_HT = (const float*)d_in[1];   // [EE x NN]
    const float* W        = (const float*)d_in[2];   // [EE]
    float* G              = (float*)d_out;           // [NN x NN]

    float* dvec         = (float*)d_ws;
    unsigned short* Abf = (unsigned short*)((char*)d_ws + 32768);
    unsigned short* Bbf = Abf + (size_t)NN * EE;

    degree_rsqrt_kernel<<<NN, 256, 0, stream>>>(H, W, dvec);
    scaleA_kernel<<<4096, 256, 0, stream>>>(H, dvec, Abf);
    scaleB_kernel<<<4096, 256, 0, stream>>>(invDE_HT, W, dvec, Bbf);

    dim3 grid(NN / BM, NN / BN);
    gemm_wmma_bf16_kernel<<<grid, 256, 0, stream>>>(Abf, Bbf, G);
}
